// ExplaiNN_66606352827149
// MI455X (gfx1250) — matrix operand order repaired and checked by reading, not compiled, not run on hardware
//
#include <hip/hip_runtime.h>

#define B_SZ  256
#define L_SZ  1000
#define G_SZ  300
#define K_SZ  19
#define HID   100
#define P_SZ  100
#define NCLS  2
#define EPS   1e-5f

typedef __bf16 bf16_t;
typedef __attribute__((ext_vector_type(16))) __bf16 v16bf;
typedef __attribute__((ext_vector_type(8)))  __bf16 v8bf;
typedef __attribute__((ext_vector_type(4)))  __bf16 v4bf;
typedef __attribute__((ext_vector_type(8)))  float  v8f;

#define GT_TILES   19     // ceil(300/16) group tiles
#define PB_BLOCKS  13     // ceil(1000/80) position blocks (80 = lcm(16,10))
#define POOL_STRIDE 128   // P=100 padded to 128 (K dim of stage-2 GEMM)
#define HPAD        112   // HID=100 padded to 7*16
#define KPAD        128
#define PROWS       84    // pooling stage row stride (bank-conflict padding)

// interleaved x in LDS (bf16): elem m = (pos+9)*4 + c ; B-frag elem = n*4 + k
#define XELEMS 4256       // covers pos in [-9, 1054]; valid data m in [36,4036)

#define POOLED_ELEMS ((size_t)G_SZ * B_SZ * POOL_STRIDE)  // bf16
#define W1P_ELEMS    ((size_t)G_SZ * HPAD * KPAD)         // bf16
#define POOLED_BYTES (POOLED_ELEMS * 2)
#define W1P_BYTES    (W1P_ELEMS * 2)

static __device__ __forceinline__ v16bf cat_bf(v8bf lo, v8bf hi) {
    return __builtin_shufflevector(lo, hi, 0, 1, 2, 3, 4, 5, 6, 7,
                                   8, 9, 10, 11, 12, 13, 14, 15);
}
static __device__ __forceinline__ v8bf cat4_bf(v4bf lo, v4bf hi) {
    return __builtin_shufflevector(lo, hi, 0, 1, 2, 3, 4, 5, 6, 7);
}

// ---------------------------------------------------------------------------
// Stage 1: conv(4->G,k=19,pad=9) + BN1 + ReLU + MaxPool(10), fully fused.
// grid = (19 group-tiles, 256 batches), block = 256 (8 waves).
// im2col K order is k = dk*4 + c; x is staged in LDS position-major as bf16,
// so each B-fragment is 16 CONSECUTIVE bf16 -> 4x ds_load_b64, no converts.
// k in [76,96) reads in-range garbage; the weight A-fragments are zero there.
// ---------------------------------------------------------------------------
__global__ __launch_bounds__(256) void conv_pool_kernel(
    const float* __restrict__ x, const float* __restrict__ conv_w,
    const float* __restrict__ conv_b,
    const float* __restrict__ bn1_g, const float* __restrict__ bn1_b,
    const float* __restrict__ bn1_m, const float* __restrict__ bn1_v,
    bf16_t* __restrict__ pooled)
{
    __shared__ bf16_t lds_x[XELEMS];           // interleaved [pos+9][c], bf16
    __shared__ float  lds_pool[8 * 16 * PROWS];

    const int gt   = blockIdx.x;
    const int b    = blockIdx.y;
    const int tid  = threadIdx.x;
    const int wave = tid >> 5;
    const int lane = tid & 31;
    const int half = lane >> 4;
    const int l16  = lane & 15;

    // Zero the interleaved buffer (halo + tail) with 32-bit stores.
    {
        unsigned int* z = reinterpret_cast<unsigned int*>(lds_x);
        for (int m = tid; m < XELEMS / 2; m += 256) z[m] = 0u;
    }
    __syncthreads();
    // Fill: one position per iteration, all 4 channels packed -> ds_store_b64.
    for (int j = tid; j < L_SZ; j += 256) {
        v4bf p;
#pragma unroll
        for (int c = 0; c < 4; ++c)
            p[c] = (bf16_t)x[(b * 4 + c) * L_SZ + j];
        *reinterpret_cast<v4bf*>(&lds_x[(j + 9) * 4]) = p;
    }
    __syncthreads();

    // A fragments: conv weights with reordered k = dk*4 + c, zero for k>=76.
    const int gg = gt * 16 + l16;              // A row: M = lane%16
    v16bf afrag[3];
#pragma unroll
    for (int kc = 0; kc < 3; ++kc) {
#pragma unroll
        for (int e = 0; e < 16; ++e) {
            int k = 32 * kc + 16 * half + e;
            float v = 0.0f;
            if (k < 76 && gg < G_SZ) {
                int dk = k >> 2, c = k & 3;
                v = conv_w[(gg * 4 + c) * K_SZ + dk];
            }
            afrag[kc][e] = (bf16_t)v;
        }
    }

    // Fused conv_b + BN1 constants per C/D row r (M = r + 8*half).
    float As[8], Bs[8];
#pragma unroll
    for (int r = 0; r < 8; ++r) {
        int g = gt * 16 + r + 8 * half;
        float s = 0.0f, sh = 0.0f;
        if (g < G_SZ) {
            s  = bn1_g[g] * rsqrtf(bn1_v[g] + EPS);
            sh = (conv_b[g] - bn1_m[g]) * s + bn1_b[g];
        }
        As[r] = s; Bs[r] = sh;
    }

    float* mypool = &lds_pool[wave * 16 * PROWS];

    for (int pb = wave; pb < PB_BLOCKS; pb += 8) {
        const int l0 = pb * 80;
#pragma unroll
        for (int t = 0; t < 5; ++t) {
            const int nbase = l0 + t * 16 + l16;        // output position (B col)
            const bf16_t* colbase = &lds_x[nbase * 4];  // im2col column start
            v8f acc = {};
#pragma unroll
            for (int kc = 0; kc < 3; ++kc) {
                // 16 consecutive bf16: K = 32*kc + 16*half + e (4x b64 loads)
                const v4bf* p = reinterpret_cast<const v4bf*>(
                    colbase + 32 * kc + 16 * half);
                v16bf bfrag = cat_bf(cat4_bf(p[0], p[1]), cat4_bf(p[2], p[3]));
                acc = __builtin_amdgcn_wmma_f32_16x16x32_bf16(
                    false, afrag[kc], false, bfrag, (short)0, acc, false, false);
            }
#pragma unroll
            for (int r = 0; r < 8; ++r) {
                float v = fmaxf(acc[r] * As[r] + Bs[r], 0.0f);   // BN1 + ReLU
                mypool[(r + 8 * half) * PROWS + t * 16 + l16] = v;
            }
        }
        // Max-pool 10 -> 8 pooled cols per row; 16 rows * 8 = 128 outs, 4/lane.
#pragma unroll
        for (int q = 0; q < 4; ++q) {
            int o   = lane + 32 * q;
            int row = o >> 3, pc = o & 7;
            float m = 0.0f;                         // post-ReLU values >= 0
#pragma unroll
            for (int j = 0; j < 10; ++j)
                m = fmaxf(m, mypool[row * PROWS + pc * 10 + j]);
            int g = gt * 16 + row;
            int i = pb * 8 + pc;
            if (g < G_SZ && i < P_SZ)
                pooled[((size_t)g * B_SZ + b) * POOL_STRIDE + i] = (bf16_t)m;
        }
    }
}

// ---------------------------------------------------------------------------
// Prep: W1 (f32 [G][HID][P]) -> bf16 zero-padded [G][112][128] B layout.
// ---------------------------------------------------------------------------
__global__ __launch_bounds__(256) void w1_prep_kernel(
    const float* __restrict__ W1, bf16_t* __restrict__ w1p)
{
    size_t idx = (size_t)blockIdx.x * 256 + threadIdx.x;
    if (idx >= W1P_ELEMS) return;
    int i = (int)(idx & (KPAD - 1));
    int h = (int)((idx >> 7) % HPAD);
    int g = (int)(idx / (HPAD * KPAD));
    float v = (h < HID && i < P_SZ) ? W1[((size_t)g * HID + h) * P_SZ + i] : 0.0f;
    w1p[idx] = (bf16_t)v;
}

// ---------------------------------------------------------------------------
// Stage 2: per-group GEMM [256x100x100] + BN2 + ReLU + dot(W2) + BN3 + ReLU.
// grid = 300 (one per group), block = 256 (8 waves, 2 M-tiles each).
// ---------------------------------------------------------------------------
__global__ __launch_bounds__(256) void group_gemm_kernel(
    const bf16_t* __restrict__ pooled, const bf16_t* __restrict__ w1p,
    const float* __restrict__ b1,
    const float* __restrict__ bn2_g, const float* __restrict__ bn2_b,
    const float* __restrict__ bn2_m, const float* __restrict__ bn2_v,
    const float* __restrict__ W2, const float* __restrict__ b2,
    const float* __restrict__ bn3_g, const float* __restrict__ bn3_b,
    const float* __restrict__ bn3_m, const float* __restrict__ bn3_v,
    float* __restrict__ h2out)
{
    __shared__ float h2acc[B_SZ];
    const int g    = blockIdx.x;
    const int tid  = threadIdx.x;
    const int wave = tid >> 5;
    const int lane = tid & 31;
    const int half = lane >> 4;
    const int l16  = lane & 15;

    h2acc[tid] = 0.0f;
    __syncthreads();

#pragma unroll
    for (int mt = 0; mt < 2; ++mt) {
        const int m0 = (wave * 2 + mt) * 16;
        const int b  = m0 + l16;

        // A fragments: pooled[b] rows (two b128 loads per K-chunk).
        v16bf A[4];
#pragma unroll
        for (int kc = 0; kc < 4; ++kc) {
            const bf16_t* p0 =
                pooled + ((size_t)g * B_SZ + b) * POOL_STRIDE + 32 * kc + 8 * half;
            v8bf lo = *reinterpret_cast<const v8bf*>(p0);
            v8bf hi = *reinterpret_cast<const v8bf*>(p0 + 16);
            A[kc] = cat_bf(lo, hi);
        }

#pragma unroll
        for (int nt = 0; nt < 7; ++nt) {
            const int h = nt * 16 + l16;
            v8f acc = {};
#pragma unroll
            for (int kc = 0; kc < 4; ++kc) {
                const bf16_t* q =
                    w1p + ((size_t)g * HPAD + h) * KPAD + 32 * kc + 8 * half;
                v8bf lo = *reinterpret_cast<const v8bf*>(q);
                v8bf hi = *reinterpret_cast<const v8bf*>(q + 16);
                acc = __builtin_amdgcn_wmma_f32_16x16x32_bf16(
                    false, A[kc], false, cat_bf(lo, hi), (short)0, acc,
                    false, false);
            }
            // Fused: +b1, BN2, ReLU, * W2[g,h]; padded h -> w2v = 0 kills garbage.
            float s2 = 0.0f, C2 = 0.0f, w2v = 0.0f;
            if (h < HID) {
                int idx = g * HID + h;
                s2  = bn2_g[idx] * rsqrtf(bn2_v[idx] + EPS);
                C2  = (b1[idx] - bn2_m[idx]) * s2 + bn2_b[idx];
                w2v = W2[idx];
            }
#pragma unroll
            for (int r = 0; r < 8; ++r) {
                float v = fmaxf(acc[r] * s2 + C2, 0.0f);
                atomicAdd(&h2acc[m0 + r + 8 * half], v * w2v);
            }
        }
    }
    __syncthreads();

    // +b2, BN3, ReLU -> h2[b][g]
    {
        float v  = h2acc[tid] + b2[g];
        float s3 = bn3_g[g] * rsqrtf(bn3_v[g] + EPS);
        v = fmaxf((v - bn3_m[g]) * s3 + bn3_b[g], 0.0f);
        h2out[(size_t)tid * G_SZ + g] = v;
    }
}

// ---------------------------------------------------------------------------
// Stage 3: classifier [256x300] @ [300x2]^T + bias.
// ---------------------------------------------------------------------------
__global__ __launch_bounds__(256) void cls_kernel(
    const float* __restrict__ h2, const float* __restrict__ cls_w,
    const float* __restrict__ cls_b, float* __restrict__ out)
{
    int b = threadIdx.x;
    float a0 = cls_b[0], a1 = cls_b[1];
    for (int g = 0; g < G_SZ; ++g) {
        float v = h2[(size_t)b * G_SZ + g];
        a0 += v * cls_w[g];
        a1 += v * cls_w[G_SZ + g];
    }
    out[b * NCLS + 0] = a0;
    out[b * NCLS + 1] = a1;
}

// ---------------------------------------------------------------------------
extern "C" void kernel_launch(void* const* d_in, const int* in_sizes, int n_in,
                              void* d_out, int out_size, void* d_ws, size_t ws_size,
                              hipStream_t stream)
{
    (void)in_sizes; (void)n_in; (void)out_size; (void)ws_size;

    const float* x      = (const float*)d_in[0];
    const float* conv_w = (const float*)d_in[1];
    const float* conv_b = (const float*)d_in[2];
    const float* bn1_g  = (const float*)d_in[3];
    const float* bn1_b  = (const float*)d_in[4];
    const float* bn1_m  = (const float*)d_in[5];
    const float* bn1_v  = (const float*)d_in[6];
    const float* W1     = (const float*)d_in[7];
    const float* b1     = (const float*)d_in[8];
    const float* bn2_g  = (const float*)d_in[9];
    const float* bn2_b  = (const float*)d_in[10];
    const float* bn2_m  = (const float*)d_in[11];
    const float* bn2_v  = (const float*)d_in[12];
    const float* W2     = (const float*)d_in[13];
    const float* b2     = (const float*)d_in[14];
    const float* bn3_g  = (const float*)d_in[15];
    const float* bn3_b  = (const float*)d_in[16];
    const float* bn3_m  = (const float*)d_in[17];
    const float* bn3_v  = (const float*)d_in[18];
    const float* cls_w  = (const float*)d_in[19];
    const float* cls_b  = (const float*)d_in[20];

    char*   ws     = (char*)d_ws;
    bf16_t* pooled = (bf16_t*)ws;
    bf16_t* w1p    = (bf16_t*)(ws + POOLED_BYTES);
    float*  h2w    = (float*)(ws + POOLED_BYTES + W1P_BYTES);

    // Zero K-padding of pooled buffer (bf16 zero == 0x0000).
    hipMemsetAsync(pooled, 0, POOLED_BYTES, stream);

    // Convert W1 to padded bf16 WMMA-B layout.
    int prep_blocks = (int)((W1P_ELEMS + 255) / 256);
    w1_prep_kernel<<<prep_blocks, 256, 0, stream>>>(W1, w1p);

    // Fused conv + BN1 + ReLU + maxpool (WMMA).
    conv_pool_kernel<<<dim3(GT_TILES, B_SZ), 256, 0, stream>>>(
        x, conv_w, conv_b, bn1_g, bn1_b, bn1_m, bn1_v, pooled);

    // Per-group GEMM + BN2/ReLU + W2 reduction + BN3/ReLU (WMMA).
    group_gemm_kernel<<<G_SZ, 256, 0, stream>>>(
        pooled, w1p, b1, bn2_g, bn2_b, bn2_m, bn2_v,
        W2, b2, bn3_g, bn3_b, bn3_m, bn3_v, h2w);

    // Final classifier.
    cls_kernel<<<1, 256, 0, stream>>>(h2w, cls_w, cls_b, (float*)d_out);
}
